// LocalRelationBlock_47734266527900
// MI455X (gfx1250) — matrix operand order repaired and checked by reading, not compiled, not run on hardware
//
#include <hip/hip_runtime.h>
#include <hip/hip_bf16.h>

#define BATCH 8192
#define CCH   256
#define HW    25
#define NPAD  32
#define SEH   64

typedef __attribute__((ext_vector_type(16))) __bf16 v16bf;
typedef __attribute__((ext_vector_type(8)))  float  v8f;

union Frag16 { v16bf v; uint4 q[2]; };

__device__ __forceinline__ unsigned short f2bf(float f) {
    // round-to-nearest-even f32 -> bf16
    unsigned int u = __float_as_uint(f);
    unsigned int r = u + 0x7FFFu + ((u >> 16) & 1u);
    return (unsigned short)(r >> 16);
}

__device__ __forceinline__ float gelu_exact(float x) {
    return 0.5f * x * (1.0f + erff(x * 0.70710678118654752440f));
}

__device__ __forceinline__ v8f wmma_bf16(const Frag16& a, const Frag16& b, v8f c) {
    // (neg_a, A, neg_b, B, c_mod, C, reuse_a, reuse_b)
    return __builtin_amdgcn_wmma_f32_16x16x32_bf16(false, a.v, false, b.v,
                                                   (short)0, c, false, false);
}

// ---------------------------------------------------------------------------
// Kernel 0: convert W1 [256,256] and W2 [256,256,3,3] (tap-sliced) to bf16
// ---------------------------------------------------------------------------
__global__ void prep_weights(const float* __restrict__ W1,
                             const float* __restrict__ W2,
                             unsigned short* __restrict__ w1bf,
                             unsigned short* __restrict__ w2bf) {
    int idx = blockIdx.x * 256 + threadIdx.x;
    if (idx < CCH * CCH) {
        w1bf[idx] = f2bf(W1[idx]);
    }
    int idx2 = idx - CCH * CCH;
    if (idx2 >= 0 && idx2 < 9 * CCH * CCH) {
        int t = idx2 / (CCH * CCH);
        int rem = idx2 - t * (CCH * CCH);
        int o = rem >> 8, c = rem & 255;
        // w2bf[(t*256+o)*256 + c] = W2[o][c][ky][kx], t = ky*3+kx
        w2bf[idx2] = f2bf(W2[(o * CCH + c) * 9 + t]);
    }
}

// ---------------------------------------------------------------------------
// Kernel 1: 1x1 conv (GEMM) + GroupNorm + GELU -> bf16 activation
// One workgroup (256 threads = 8 wave32) per batch element.
// ---------------------------------------------------------------------------
__global__ void __launch_bounds__(256)
conv1_gn_gelu(const float* __restrict__ x,
              const unsigned short* __restrict__ w1bf,
              const float* __restrict__ g1, const float* __restrict__ b1,
              unsigned short* __restrict__ y1) {
    __shared__ __align__(16) unsigned short xT[NPAD * CCH];  // [n][c] bf16
    __shared__ __align__(16) float outbuf[CCH * NPAD];       // [o][n] f32
    __shared__ float chs[CCH], chq[CCH];
    __shared__ float gmean[8], grstd[8];

    const int b    = blockIdx.x;
    const int tid  = threadIdx.x;
    const int lane = tid & 31;
    const int wave = tid >> 5;
    const int hi   = lane >> 4;
    const int l15  = lane & 15;

    // stage x_b -> xT bf16, transposed [n][c], cols 25..31 zero
    {
        int n  = tid >> 3;
        int c0 = (tid & 7) * 32;
        const float* xb = x + (size_t)b * CCH * HW;
        for (int i = 0; i < 32; ++i) {
            int c = c0 + i;
            float v = (n < HW) ? xb[c * HW + n] : 0.0f;
            xT[n * CCH + c] = f2bf(v);
        }
    }
    __syncthreads();

    // GEMM: out[o][n] = sum_c W1[o][c] * x[c][n]; M=256,N=32,K=256
    for (int mi = 0; mi < 2; ++mi) {
        int mt = wave * 2 + mi;
        const unsigned short* arow = w1bf + (mt * 16 + l15) * CCH;
        __builtin_prefetch(arow, 0, 0);
        v8f acc0 = {}, acc1 = {};
        for (int kk = 0; kk < 8; ++kk) {
            int cbase = kk * 32 + hi * 16;                // A: 16 contiguous K
            Frag16 af;
            af.q[0] = *(const uint4*)(arow + cbase);
            af.q[1] = *(const uint4*)(arow + cbase + 8);
            int kb = kk * 32 + hi * 16;                   // B: 16 contiguous K
            Frag16 bf0, bf1;
            const uint4* p0 = (const uint4*)(&xT[l15 * CCH + kb]);
            bf0.q[0] = p0[0]; bf0.q[1] = p0[1];
            const uint4* p1 = (const uint4*)(&xT[(16 + l15) * CCH + kb]);
            bf1.q[0] = p1[0]; bf1.q[1] = p1[1];
            acc0 = wmma_bf16(af, bf0, acc0);
            acc1 = wmma_bf16(af, bf1, acc1);
        }
        for (int r = 0; r < 8; ++r) {
            int oo = mt * 16 + r + 8 * hi;                // C/D layout
            outbuf[oo * NPAD + l15]      = acc0[r];
            outbuf[oo * NPAD + 16 + l15] = acc1[r];
        }
    }
    __syncthreads();

    // GroupNorm: 8 groups of (32 ch x 25 spatial) = 800 values each
    {
        float s = 0.f, s2 = 0.f;
        for (int n = 0; n < HW; ++n) {
            float v = outbuf[tid * NPAD + n];
            s += v; s2 += v * v;
        }
        chs[tid] = s; chq[tid] = s2;
    }
    __syncthreads();
    if (tid < 8) {
        float a = 0.f, q = 0.f;
        for (int i = 0; i < 32; ++i) { a += chs[tid * 32 + i]; q += chq[tid * 32 + i]; }
        float mean = a * (1.0f / 800.0f);
        float var  = q * (1.0f / 800.0f) - mean * mean;
        gmean[tid] = mean;
        grstd[tid] = rsqrtf(var + 1e-5f);
    }
    __syncthreads();
    {
        float m  = gmean[tid >> 5], rs = grstd[tid >> 5];
        float ga = g1[tid], be = b1[tid];
        unsigned short* yrow = y1 + ((size_t)b * CCH + tid) * HW;
        for (int n = 0; n < HW; ++n) {
            float v = (outbuf[tid * NPAD + n] - m) * rs * ga + be;
            yrow[n] = f2bf(gelu_exact(v));
        }
    }
}

// ---------------------------------------------------------------------------
// Kernel 2: 3x3 conv as 9 shifted GEMMs + GroupNorm + GELU + residual -> f32
// Writes the pre-gate activation straight into d_out's output region.
// ---------------------------------------------------------------------------
__global__ void __launch_bounds__(256)
conv2_gn_gelu_res(const unsigned short* __restrict__ y1,
                  const unsigned short* __restrict__ w2bf,
                  const float* __restrict__ g2, const float* __restrict__ b2,
                  const float* __restrict__ x,
                  float* __restrict__ outp) {
    __shared__ __align__(16) unsigned short yT[NPAD * CCH];  // [n][c] bf16
    __shared__ __align__(16) float outbuf[CCH * NPAD];
    __shared__ float chs[CCH], chq[CCH];
    __shared__ float gmean[8], grstd[8];

    const int b    = blockIdx.x;
    const int tid  = threadIdx.x;
    const int lane = tid & 31;
    const int wave = tid >> 5;
    const int hi   = lane >> 4;
    const int l15  = lane & 15;

    v8f acc[2][2] = {};
    const unsigned short* yb = y1 + (size_t)b * CCH * HW;

    for (int tap = 0; tap < 9; ++tap) {
        int dy = tap / 3 - 1, dx = tap % 3 - 1;
        __syncthreads();   // previous tap's fragment reads complete
        {
            int n  = tid >> 3;
            int c0 = (tid & 7) * 32;
            int h  = n / 5, w = n % 5;
            int hh = h + dy, ww = w + dx;
            bool ok = (n < HW) && (hh >= 0) && (hh < 5) && (ww >= 0) && (ww < 5);
            int sp = hh * 5 + ww;
            for (int i = 0; i < 32; ++i) {
                int c = c0 + i;
                yT[n * CCH + c] = ok ? yb[c * HW + sp] : (unsigned short)0;
            }
        }
        __syncthreads();
        const unsigned short* wt = w2bf + tap * (CCH * CCH);
        for (int mi = 0; mi < 2; ++mi) {
            int mt = wave * 2 + mi;
            const unsigned short* arow = wt + (mt * 16 + l15) * CCH;
            for (int kk = 0; kk < 8; ++kk) {
                int cbase = kk * 32 + hi * 16;
                Frag16 af;
                af.q[0] = *(const uint4*)(arow + cbase);
                af.q[1] = *(const uint4*)(arow + cbase + 8);
                int kb = kk * 32 + hi * 16;
                Frag16 bf0, bf1;
                const uint4* p0 = (const uint4*)(&yT[l15 * CCH + kb]);
                bf0.q[0] = p0[0]; bf0.q[1] = p0[1];
                const uint4* p1 = (const uint4*)(&yT[(16 + l15) * CCH + kb]);
                bf1.q[0] = p1[0]; bf1.q[1] = p1[1];
                acc[mi][0] = wmma_bf16(af, bf0, acc[mi][0]);
                acc[mi][1] = wmma_bf16(af, bf1, acc[mi][1]);
            }
        }
    }
    __syncthreads();
    for (int mi = 0; mi < 2; ++mi) {
        int mt = wave * 2 + mi;
        for (int r = 0; r < 8; ++r) {
            int oo = mt * 16 + r + 8 * hi;
            outbuf[oo * NPAD + l15]      = acc[mi][0][r];
            outbuf[oo * NPAD + 16 + l15] = acc[mi][1][r];
        }
    }
    __syncthreads();

    {
        float s = 0.f, s2 = 0.f;
        for (int n = 0; n < HW; ++n) {
            float v = outbuf[tid * NPAD + n];
            s += v; s2 += v * v;
        }
        chs[tid] = s; chq[tid] = s2;
    }
    __syncthreads();
    if (tid < 8) {
        float a = 0.f, q = 0.f;
        for (int i = 0; i < 32; ++i) { a += chs[tid * 32 + i]; q += chq[tid * 32 + i]; }
        float mean = a * (1.0f / 800.0f);
        float var  = q * (1.0f / 800.0f) - mean * mean;
        gmean[tid] = mean;
        grstd[tid] = rsqrtf(var + 1e-5f);
    }
    __syncthreads();
    {
        float m  = gmean[tid >> 5], rs = grstd[tid >> 5];
        float ga = g2[tid], be = b2[tid];
        const float* xr = x    + ((size_t)b * CCH + tid) * HW;
        float*       orow = outp + ((size_t)b * CCH + tid) * HW;
        for (int n = 0; n < HW; ++n) {
            float v = (outbuf[tid * NPAD + n] - m) * rs * ga + be;
            orow[n] = gelu_exact(v) + xr[n];
        }
    }
}

// ---------------------------------------------------------------------------
// Kernel 3: masked SE pooling + fc1/GELU + fc2/sigmoid; writes gate and
// rescales the output in place (out *= gate * mask * mask).
// ---------------------------------------------------------------------------
__global__ void __launch_bounds__(256)
se_gate_scale(float* __restrict__ out,
              const float* __restrict__ pm,
              const float* __restrict__ fc1w, const float* __restrict__ fc1b,
              const float* __restrict__ fc2w, const float* __restrict__ fc2b,
              float* __restrict__ gate_out) {
    __shared__ float maskv[32];
    __shared__ float pooled[CCH];
    __shared__ float hbuf[SEH];
    __shared__ float denom_s;

    const int b   = blockIdx.x;
    const int tid = threadIdx.x;

    if (tid < 32) maskv[tid] = (tid < HW) ? pm[b * HW + tid] : 0.0f;
    __syncthreads();
    if (tid == 0) {
        float s = 0.f;
        for (int p = 0; p < HW; ++p) s += maskv[p];
        denom_s = fmaxf(s, 1e-8f);
    }
    __syncthreads();
    {
        const float* row = out + ((size_t)b * CCH + tid) * HW;
        float s = 0.f;
        for (int p = 0; p < HW; ++p) s += row[p] * maskv[p];
        pooled[tid] = s / denom_s;
    }
    __syncthreads();
    if (tid < SEH) {
        const float* wr = fc1w + tid * CCH;
        float a = fc1b[tid];
        for (int c = 0; c < CCH; ++c) a += pooled[c] * wr[c];
        hbuf[tid] = gelu_exact(a);
    }
    __syncthreads();
    {
        const float* wr = fc2w + tid * SEH;
        float a = fc2b[tid];
        for (int j = 0; j < SEH; ++j) a += hbuf[j] * wr[j];
        float g = 1.0f / (1.0f + expf(-a));
        gate_out[(size_t)b * CCH + tid] = g;
        float* row = out + ((size_t)b * CCH + tid) * HW;
        for (int p = 0; p < HW; ++p) {
            float mv = maskv[p];
            row[p] = row[p] * g * mv * mv;
        }
    }
}

// ---------------------------------------------------------------------------
extern "C" void kernel_launch(void* const* d_in, const int* in_sizes, int n_in,
                              void* d_out, int out_size, void* d_ws, size_t ws_size,
                              hipStream_t stream) {
    const float* x   = (const float*)d_in[0];
    const float* pm  = (const float*)d_in[1];
    const float* W1  = (const float*)d_in[2];
    const float* g1  = (const float*)d_in[3];
    const float* b1  = (const float*)d_in[4];
    const float* W2  = (const float*)d_in[5];
    const float* g2  = (const float*)d_in[6];
    const float* b2  = (const float*)d_in[7];
    const float* f1w = (const float*)d_in[8];
    const float* f1b = (const float*)d_in[9];
    const float* f2w = (const float*)d_in[10];
    const float* f2b = (const float*)d_in[11];

    char* ws = (char*)d_ws;
    unsigned short* w1bf = (unsigned short*)(ws);                       // 128 KB
    unsigned short* w2bf = (unsigned short*)(ws + 131072);              // 1.125 MB
    unsigned short* y1   = (unsigned short*)(ws + 131072 + 1179648);    // 100 MB

    float* out  = (float*)d_out;                                        // [B,256,25]
    float* gate = out + (size_t)BATCH * CCH * HW;                       // [B,256]

    prep_weights<<<(CCH * CCH * 10 + 255) / 256, 256, 0, stream>>>(W1, W2, w1bf, w2bf);
    conv1_gn_gelu<<<BATCH, 256, 0, stream>>>(x, w1bf, g1, b1, y1);
    conv2_gn_gelu_res<<<BATCH, 256, 0, stream>>>(y1, w2bf, g2, b2, x, out);
    se_gate_scale<<<BATCH, 256, 0, stream>>>(out, pm, f1w, f1b, f2w, f2b, gate);
}
